// AdditiveAttention_81226421502318
// MI455X (gfx1250) — matrix operand order repaired and checked by reading, not compiled, not run on hardware
//
#include <hip/hip_runtime.h>

typedef __attribute__((ext_vector_type(16))) _Float16 v16h;
typedef __attribute__((ext_vector_type(2)))  _Float16 v2h;
typedef __attribute__((ext_vector_type(8)))  float    v8f;

#define BATCH 2
#define TQ    1024
#define TK    1024
#define DD    64
#define QTILE 16
#define WAVES 8
#define SPLIT 4                   // split-K factor over Tk
#define KS    (TK / SPLIT)        // 256 keys per block
#define KW    (KS / WAVES)        // 32 keys per wave (one WMMA K-step)
#define NTILES 128                // BATCH * TQ / QTILE
#define REC_STRIDE (QTILE * DD + 2 * QTILE)  // pacc[16][64] + pmax[16] + psum[16]

__device__ __forceinline__ float fast_tanh(float x) {
#if __has_builtin(__builtin_amdgcn_tanhf)
  return __builtin_amdgcn_tanhf(x);          // v_tanh_f32 on gfx1250
#else
  float e = __expf(2.0f * x);                // tanh(x) = 1 - 2/(e^{2x}+1)
  return 1.0f - 2.0f / (e + 1.0f);
#endif
}

struct Smem {
  float    qs[QTILE][DD];        //  4 KB   query tile
  float    sc[DD];               //  256 B  scale vector
  float    s[QTILE][KS];         // 16 KB   raw scores (this split's 256 keys)
  _Float16 p[QTILE][KS];         //  8 KB   exp(s - localmax) f16 (WMMA A)
  float    red[QTILE][16];       //  1 KB   softmax partial reductions
  float    rowmax[QTILE];
  float    rowsum[QTILE];
  float    outAcc[QTILE][DD];    //  4 KB   cross-wave f32 accumulator
};

// ---------------------------------------------------------------------------
// Kernel 1: one block = one (b, q-tile, k-split).  Computes local softmax
// stats and the unnormalized partial P@V via v_wmma_f32_16x16x32_f16.
// ---------------------------------------------------------------------------
__global__ __launch_bounds__(256)
void AdditiveAttention_partial_kernel(const float* __restrict__ query,
                                      const float* __restrict__ value,
                                      const float* __restrict__ scale,
                                      float* __restrict__ ws) {
  __shared__ Smem sm;
  const int t     = threadIdx.x;
  const int wave  = t >> 5;
  const int lane  = t & 31;
  const int tile  = blockIdx.x >> 2;         // 0..127
  const int split = blockIdx.x & (SPLIT - 1);
  const int b     = tile >> 6;
  const int q0    = (tile & 63) * QTILE;
  const int kb    = split * KS;              // first key of this split

  const float* qb = query + ((size_t)b * TQ + q0) * DD;
  const float* vb = value + (size_t)b * TK * DD;

  // ---- Phase 0: stage Q tile + scale, zero accumulator -------------------
  #pragma unroll
  for (int i = 0; i < 4; ++i) {
    int idx = t + 256 * i;                   // 1024 = 16*64 elements
    sm.qs[idx >> 6][idx & 63]     = qb[idx];
    sm.outAcc[idx >> 6][idx & 63] = 0.0f;
  }
  if (t < DD) sm.sc[t] = scale[t];
  __syncthreads();

  // ---- Phase 1: scores s[m][k] = sum_d sc[d]*tanh(q[m][d]+v[k][d]) -------
  {
    const int m    = lane & 15;              // q-row owned by this lane
    const int ksub = lane >> 4;              // k parity
    float qrow[DD];
    #pragma unroll
    for (int d = 0; d < DD; ++d) qrow[d] = sm.qs[m][d];
    const int kw = wave * KW;                // local key offset of this wave
    for (int j = 0; j < KW / 2; ++j) {       // 16 keys per lane
      const int kl = kw + 2 * j + ksub;      // local key index [0,256)
      const float* vrow = vb + (size_t)(kb + kl) * DD;
      float acc = 0.0f;
      #pragma unroll
      for (int d = 0; d < DD; ++d)
        acc += sm.sc[d] * fast_tanh(qrow[d] + vrow[d]);
      sm.s[m][kl] = acc;
    }
  }
  __syncthreads();

  // ---- Phase 2: local softmax stats; exp(s - localmax) -> f16 P ----------
  {
    const int r    = t >> 4;                 // row 0..15
    const int seg  = t & 15;                 // 16 segments of 16 keys
    const int base = seg * (KS / 16);
    float lmax = -3.402823e38f;
    #pragma unroll
    for (int i = 0; i < KS / 16; ++i) lmax = fmaxf(lmax, sm.s[r][base + i]);
    sm.red[r][seg] = lmax;
    __syncthreads();
    if (t < QTILE) {
      float mx = sm.red[t][0];
      #pragma unroll
      for (int i = 1; i < 16; ++i) mx = fmaxf(mx, sm.red[t][i]);
      sm.rowmax[t] = mx;
    }
    __syncthreads();
    const float rm = sm.rowmax[r];
    float lsum = 0.0f;
    #pragma unroll
    for (int i = 0; i < KS / 16; ++i) {
      float e = __expf(sm.s[r][base + i] - rm);
      sm.p[r][base + i] = (_Float16)e;
      lsum += e;
    }
    sm.red[r][seg] = lsum;
    __syncthreads();
    if (t < QTILE) {
      float sum = 0.0f;
      #pragma unroll
      for (int i = 0; i < 16; ++i) sum += sm.red[t][i];
      sm.rowsum[t] = sum;
    }
    __syncthreads();
  }

  // ---- Phase 3: per-wave P[16x32] @ V[32x64] via v_wmma_f32_16x16x32_f16 -
  {
    const int m    = lane & 15;              // A row / B column within tile
    const int half = lane >> 4;
    const int k0   = wave * KW;              // this wave's local K=32 chunk
    // A fragment (16x32 f16): lane<16 holds K {0..7,16..23}; lane>=16: +8
    v16h a;
    const int kbaseA = half * 8;
    const unsigned int* prow = (const unsigned int*)&sm.p[m][0];
    #pragma unroll
    for (int i = 0; i < 8; ++i) {
      const int kp = (i < 4) ? (kbaseA + 2 * i) : (16 + kbaseA + 2 * (i - 4));
      unsigned int u = prow[(k0 + kp) >> 1];     // two contiguous f16
      v2h pr = __builtin_bit_cast(v2h, u);
      a[2 * i]     = pr[0];
      a[2 * i + 1] = pr[1];
    }
    // B fragments (32x16 f16): lane = column N; lane<16 holds K {0..7,16..23}; lane>=16: +8
    const int koffB = half * 8;
    v8f acc[4] = {};
    #pragma unroll
    for (int nt = 0; nt < 4; ++nt) {
      const int d0 = nt * 16 + m;
      v16h bf;
      #pragma unroll
      for (int j2 = 0; j2 < 8; ++j2) {
        const int k = kb + k0 + koffB + 2 * j2 + ((j2 >= 4) ? 8 : 0);
        bf[2 * j2]     = (_Float16)vb[(size_t)k * DD + d0];
        bf[2 * j2 + 1] = (_Float16)vb[(size_t)(k + 1) * DD + d0];
      }
      acc[nt] = __builtin_amdgcn_wmma_f32_16x16x32_f16(
          /*neg_a=*/false, a, /*neg_b=*/false, bf,
          /*c_mod=*/(short)0, acc[nt], /*reuse_a=*/false, /*reuse_b=*/false);
    }
    // C layout: VGPR r holds M = r + 8*(lane/16), N = lane%16
    #pragma unroll
    for (int r = 0; r < 8; ++r) {
      const int mm = r + 8 * half;
      atomicAdd(&sm.outAcc[mm][ 0 + m], acc[0][r]);
      atomicAdd(&sm.outAcc[mm][16 + m], acc[1][r]);
      atomicAdd(&sm.outAcc[mm][32 + m], acc[2][r]);
      atomicAdd(&sm.outAcc[mm][48 + m], acc[3][r]);
    }
  }
  __syncthreads();

  // ---- Write partial record: pacc[16][64], pmax[16], psum[16] ------------
  float* rec = ws + (size_t)blockIdx.x * REC_STRIDE;
  #pragma unroll
  for (int i = 0; i < 4; ++i) {
    int idx = t + 256 * i;
    rec[idx] = sm.outAcc[idx >> 6][idx & 63];
  }
  if (t < QTILE) {
    rec[QTILE * DD + t]         = sm.rowmax[t];
    rec[QTILE * DD + QTILE + t] = sm.rowsum[t];
  }
}

// ---------------------------------------------------------------------------
// Kernel 2: combine the SPLIT partials per (b, q-tile) with max-rescaling.
// out[m][d] = sum_s pacc_s[m][d]*exp(pmax_s-M) / sum_s psum_s*exp(pmax_s-M)
// ---------------------------------------------------------------------------
__global__ __launch_bounds__(256)
void AdditiveAttention_combine_kernel(const float* __restrict__ ws,
                                      float* __restrict__ out) {
  __shared__ float wgt[SPLIT][QTILE];
  const int t    = threadIdx.x;
  const int tile = blockIdx.x;
  const int b    = tile >> 6;
  const int q0   = (tile & 63) * QTILE;
  const float* recs = ws + (size_t)tile * SPLIT * REC_STRIDE;

  if (t < QTILE) {
    float pm[SPLIT], ps[SPLIT];
    #pragma unroll
    for (int s = 0; s < SPLIT; ++s) {
      const float* rec = recs + (size_t)s * REC_STRIDE;
      pm[s] = rec[QTILE * DD + t];
      ps[s] = rec[QTILE * DD + QTILE + t];
    }
    float M = pm[0];
    #pragma unroll
    for (int s = 1; s < SPLIT; ++s) M = fmaxf(M, pm[s]);
    float denom = 0.0f;
    float e[SPLIT];
    #pragma unroll
    for (int s = 0; s < SPLIT; ++s) {
      e[s] = __expf(pm[s] - M);
      denom += ps[s] * e[s];
    }
    const float rinv = 1.0f / denom;
    #pragma unroll
    for (int s = 0; s < SPLIT; ++s) wgt[s][t] = e[s] * rinv;
  }
  __syncthreads();

  float* ob = out + ((size_t)b * TQ + q0) * DD;
  #pragma unroll
  for (int i = 0; i < 4; ++i) {
    int idx = t + 256 * i;                   // 1024 = 16*64 elements
    int m = idx >> 6;
    float o = 0.0f;
    #pragma unroll
    for (int s = 0; s < SPLIT; ++s)
      o += recs[(size_t)s * REC_STRIDE + idx] * wgt[s][m];
    ob[idx] = o;
  }
}

extern "C" void kernel_launch(void* const* d_in, const int* in_sizes, int n_in,
                              void* d_out, int out_size, void* d_ws, size_t ws_size,
                              hipStream_t stream) {
  (void)in_sizes; (void)n_in; (void)out_size; (void)ws_size;
  const float* query = (const float*)d_in[0];
  const float* value = (const float*)d_in[1];
  const float* scale = (const float*)d_in[2];
  float* out = (float*)d_out;
  float* ws  = (float*)d_ws;   // NTILES*SPLIT*REC_STRIDE floats ~= 2.1 MB

  AdditiveAttention_partial_kernel<<<dim3(NTILES * SPLIT), dim3(256), 0, stream>>>(
      query, value, scale, ws);
  AdditiveAttention_combine_kernel<<<dim3(NTILES), dim3(256), 0, stream>>>(
      ws, out);
}
